// FirstEncoder_75118978007403
// MI455X (gfx1250) — compile-verified, matrix operand order checked by reading, not run on hardware
//
#include <hip/hip_runtime.h>

// ---------------------------------------------------------------------------
// CDNA5 (gfx1250) wave32 WMMA implementation of the transformer-encoder+VAE
// reference. All matmuls run on v_wmma_f32_16x16x32_bf16 (f32 accum); tile
// movement global->LDS uses the Tensor Data Mover (tensor_load_to_lds,
// TENSORcnt) with the D# pad feature generating the bank-conflict-free
// padded LDS layout directly. Double-buffered so TDM overlaps WMMA.
// ---------------------------------------------------------------------------

typedef __attribute__((ext_vector_type(16))) __bf16 v16bf;
typedef __attribute__((ext_vector_type(8)))  float  v8f;

#define DEV static __device__ __forceinline__

#if __has_builtin(__builtin_amdgcn_tensor_load_to_lds)
#define HAVE_TDM 1
#else
#define HAVE_TDM 0
#endif

constexpr int SEQ   = 512;
constexpr int DIM   = 1024;
constexpr int LAT   = 128;
constexpr int DH    = 64;
constexpr int MROWS = 16 * SEQ;          // B*S = 8192

DEV unsigned short f2bf(float x) {       // fp32 -> bf16 RNE
  unsigned int u = __float_as_uint(x);
  u += 0x7FFFu + ((u >> 16) & 1u);
  return (unsigned short)(u >> 16);
}

union FragU { v16bf v; unsigned int u[8]; };

// Load one 16x32 bf16 WMMA operand fragment from LDS (row-major, `stride`
// halfwords per row). Matches the documented 16-bit A/B VGPR layout:
//   lane<16 : elems 0..7 = K 0..7,  elems 8..15 = K 16..23
//   lane>=16: elems 0..7 = K 8..15, elems 8..15 = K 24..31
DEV v16bf load_frag(const unsigned short* base, int stride, int lane) {
  const int m    = lane & 15;
  const int koff = (lane >> 4) << 3;     // 0 or 8
  const unsigned short* row = base + m * stride;
  FragU f;
#pragma unroll
  for (int j = 0; j < 4; ++j)
    f.u[j]     = *(const unsigned int*)(row + koff + 2 * j);
#pragma unroll
  for (int j = 0; j < 4; ++j)
    f.u[4 + j] = *(const unsigned int*)(row + 16 + koff + 2 * j);
  return f.v;
}

#if HAVE_TDM
typedef __attribute__((ext_vector_type(4))) unsigned int tdm_v4u;
typedef __attribute__((ext_vector_type(8))) int         tdm_v8i;
typedef __attribute__((ext_vector_type(4))) int         tdm_v4i;

DEV unsigned lds_addr_of(const void* p) {  // low 32 bits of generic ptr = LDS byte offset
  return (unsigned)(size_t)p;
}

// TDM 2-D tile load, bf16 elements. Writes tile_y rows of tile_x elements,
// inserting 1 DWORD of LDS padding after every 2^(pad_ivl+1) DWORDs, i.e.
// pad_ivl=3 -> row stride 34 halfwords (tile_x=32), pad_ivl=4 -> 66 (tile_x=64).
DEV void tdm_load_tile(unsigned lds_byte, const void* gsrc,
                       int tile_x, int tile_y, int stride_elems, int pad_ivl) {
  const unsigned long long ga = (unsigned long long)(size_t)gsrc;
  tdm_v4u g0;
  g0[0] = 1u;                                    // count=1: valid user D#
  g0[1] = lds_byte;                              // lds_addr
  g0[2] = (unsigned)ga;                          // global_addr[31:0]
  g0[3] = ((unsigned)(ga >> 32) & 0x01FFFFFFu)   // global_addr[56:32]
          | 0x80000000u;                         // type=2 ("image")
  tdm_v8i g1;
  g1[0] = (int)((1u << 16)                       // data_size = 2 bytes
              | (1u << 20)                       // pad_enable
              | ((unsigned)pad_ivl << 22));      // pad_interval; pad_amount=0 (1 DWORD)
  const unsigned td0 = (unsigned)stride_elems;   // tensor_dim0 (>= tile_x, no OOB)
  const unsigned td1 = 0x40000000u;              // tensor_dim1: effectively unbounded
  g1[1] = (int)((td0 & 0xFFFFu) << 16);          // bits79:48 = tensor_dim0
  g1[2] = (int)((td0 >> 16) | ((td1 & 0xFFFFu) << 16));   // + tensor_dim1 lo
  g1[3] = (int)((td1 >> 16) | ((unsigned)tile_x << 16));  // + tile_dim0
  g1[4] = (int)((unsigned)tile_y);               // tile_dim1; tile_dim2=0 (2-D)
  g1[5] = (int)(unsigned)stride_elems;           // tensor_dim0_stride[31:0]
  g1[6] = 0;                                     // stride[47:32], dim1_stride lo
  g1[7] = 0;
  const tdm_v4i gz4 = {0, 0, 0, 0};              // groups 2/3 unused (2-D tensor)
  const tdm_v8i gz8 = {0, 0, 0, 0, 0, 0, 0, 0};  // extra group (clang-23 6-arg form)
  __builtin_amdgcn_tensor_load_to_lds(g0, g1, gz4, gz4, gz8, 0);
}
#endif

DEV float block_sum(float v, float* buf) {
#pragma unroll
  for (int m = 16; m; m >>= 1) v += __shfl_xor(v, m, 32);
  if ((threadIdx.x & 31) == 0) buf[threadIdx.x >> 5] = v;
  __syncthreads();
  if (threadIdx.x == 0) {
    float s = 0.f;
#pragma unroll
    for (int i = 0; i < 8; ++i) s += buf[i];
    buf[0] = s;
  }
  __syncthreads();
  float r = buf[0];
  __syncthreads();                       // allow buf reuse by a later call
  return r;
}

// --------------------------- weight transpose ------------------------------
// Wt[n*K+k] = bf16(W[k*N+n]); all GEMM operands then share the contiguous-K
// fragment loader.
__global__ __launch_bounds__(256) void transpose_cvt(
    const float* __restrict__ W, unsigned short* __restrict__ Wt, int K, int N) {
  __shared__ float t[32][33];
  const int nb = blockIdx.x * 32, kb = blockIdx.y * 32;
  const int tx = threadIdx.x & 31, ty = threadIdx.x >> 5;   // 32 x 8
#pragma unroll
  for (int j = 0; j < 32; j += 8)
    t[ty + j][tx] = W[(size_t)(kb + ty + j) * N + nb + tx];
  __syncthreads();
#pragma unroll
  for (int j = 0; j < 32; j += 8)
    Wt[(size_t)(nb + ty + j) * K + kb + tx] = f2bf(t[tx][ty + j]);
}

// ------------------------- embed + posenc + LN1 ----------------------------
__global__ __launch_bounds__(256) void embed_ln_kernel(
    const int* __restrict__ x, const float* __restrict__ embed,
    const float* __restrict__ g, const float* __restrict__ be,
    float* __restrict__ h32, unsigned short* __restrict__ hbf) {
  __shared__ float red[8];
  const int row = blockIdx.x;
  const int s   = row & (SEQ - 1);
  const int tok = x[row];
  const float* e = embed + (size_t)tok * DIM;
  const int c0 = threadIdx.x * 4;
  float v[4];
#pragma unroll
  for (int j = 0; j < 4; ++j) {
    const int c  = c0 + j;
    const int i2 = c & ~1;
    const float div = __expf(-0.0089944730195308f * (float)i2); // ln(1e4)/D
    const float ang = (float)s * div;
    const float pe  = (c & 1) ? __cosf(ang) : __sinf(ang);
    v[j] = e[c] * 32.0f + pe;                                   // sqrt(D)=32
  }
  const float mean = block_sum(v[0] + v[1] + v[2] + v[3], red) * (1.f / DIM);
  float sq = 0.f;
#pragma unroll
  for (int j = 0; j < 4; ++j) { const float d = v[j] - mean; sq += d * d; }
  const float rstd = rsqrtf(block_sum(sq, red) * (1.f / DIM) + 1e-5f);
#pragma unroll
  for (int j = 0; j < 4; ++j) {
    const int c = c0 + j;
    const float y = (v[j] - mean) * rstd * g[c] + be[c];
    h32[(size_t)row * DIM + c] = y;
    hbf[(size_t)row * DIM + c] = f2bf(y);
  }
}

// --------------------------- residual + LN ---------------------------------
// a==b gives the reference's "f + f" doubling before LN3.
__global__ __launch_bounds__(256) void resid_ln_kernel(
    const float* __restrict__ a, const float* __restrict__ b,
    const float* __restrict__ g, const float* __restrict__ be,
    float* __restrict__ h32, unsigned short* __restrict__ hbf) {
  __shared__ float red[8];
  const int row = blockIdx.x;
  const int c0  = threadIdx.x * 4;
  const size_t base = (size_t)row * DIM;
  float v[4];
#pragma unroll
  for (int j = 0; j < 4; ++j) v[j] = a[base + c0 + j] + b[base + c0 + j];
  const float mean = block_sum(v[0] + v[1] + v[2] + v[3], red) * (1.f / DIM);
  float sq = 0.f;
#pragma unroll
  for (int j = 0; j < 4; ++j) { const float d = v[j] - mean; sq += d * d; }
  const float rstd = rsqrtf(block_sum(sq, red) * (1.f / DIM) + 1e-5f);
#pragma unroll
  for (int j = 0; j < 4; ++j) {
    const int c = c0 + j;
    const float y = (v[j] - mean) * rstd * g[c] + be[c];
    h32[base + c] = y;
    hbf[base + c] = f2bf(y);
  }
}

// ------------------------------- GEMM --------------------------------------
// C[M,N] = A[M,K](bf16) * Bt[N,K]^T(bf16) + bias, optional leaky-ReLU,
// writes fp32 and/or bf16. 128x128 tile, 8 waves, wave = 32x64 via 2x4 WMMA.
// TDM path: wave 0 DMA-loads the next K-step's A/B tiles (double-buffered,
// pad feature produces the 34-halfword padded rows) while all waves WMMA.
__global__ __launch_bounds__(256) void gemm_bf16(
    const unsigned short* __restrict__ A,
    const unsigned short* __restrict__ Bt,
    const float* __restrict__ bias,
    float* __restrict__ outF, unsigned short* __restrict__ outB,
    int M, int N, int K, int act) {
#if HAVE_TDM
  __shared__ unsigned short As[2][128][34];
  __shared__ unsigned short Bs[2][128][34];
#else
  __shared__ unsigned short As[1][128][34];
  __shared__ unsigned short Bs[1][128][34];
#endif
  const int t = threadIdx.x;
  const int lane = t & 31;
  const int w = t >> 5;
  const int mbase = blockIdx.y * 128;
  const int nbase = blockIdx.x * 128;
  const int wm = (w >> 1) * 32;       // wave M offset in tile
  const int wn = (w & 1) * 64;        // wave N offset in tile

  const v8f zero = {0.f, 0.f, 0.f, 0.f, 0.f, 0.f, 0.f, 0.f};
  v8f c[2][4];
#pragma unroll
  for (int mi = 0; mi < 2; ++mi)
#pragma unroll
    for (int ni = 0; ni < 4; ++ni) c[mi][ni] = zero;

  const int nt = K / 32;

#if HAVE_TDM
  if (w == 0) {
    tdm_load_tile(lds_addr_of(&As[0][0][0]), A  + (size_t)mbase * K, 32, 128, K, 3);
    tdm_load_tile(lds_addr_of(&Bs[0][0][0]), Bt + (size_t)nbase * K, 32, 128, K, 3);
  }
  for (int it = 0; it < nt; ++it) {
    const int buf = it & 1;
    if (w == 0) __builtin_amdgcn_s_wait_tensorcnt(0);  // tile `it` resident
    __syncthreads();
    if (w == 0 && it + 1 < nt) {                       // DMA next tile now
      const int kb2 = (it + 1) * 32;
      tdm_load_tile(lds_addr_of(&As[buf ^ 1][0][0]),
                    A  + (size_t)mbase * K + kb2, 32, 128, K, 3);
      tdm_load_tile(lds_addr_of(&Bs[buf ^ 1][0][0]),
                    Bt + (size_t)nbase * K + kb2, 32, 128, K, 3);
    }
    const v16bf a0 = load_frag(&As[buf][wm][0],      34, lane);
    const v16bf a1 = load_frag(&As[buf][wm + 16][0], 34, lane);
#pragma unroll
    for (int ni = 0; ni < 4; ++ni) {
      const v16bf bf = load_frag(&Bs[buf][wn + ni * 16][0], 34, lane);
      c[0][ni] = __builtin_amdgcn_wmma_f32_16x16x32_bf16(
          false, a0, false, bf, (short)0, c[0][ni], false, false);
      c[1][ni] = __builtin_amdgcn_wmma_f32_16x16x32_bf16(
          false, a1, false, bf, (short)0, c[1][ni], false, false);
    }
    __syncthreads();                                   // done reading `buf`
  }
#else
  const int lr = t >> 1;              // load row 0..127
  const int lc = (t & 1) * 16;        // load col 0 / 16
  for (int it = 0; it < nt; ++it) {
    const int kb = it * 32;
    union { uint4 q[2]; unsigned int d[8]; } ra, rb;
    const unsigned short* ap = A  + (size_t)(mbase + lr) * K + kb + lc;
    const unsigned short* bp = Bt + (size_t)(nbase + lr) * K + kb + lc;
    ra.q[0] = *(const uint4*)(ap);     ra.q[1] = *(const uint4*)(ap + 8);
    rb.q[0] = *(const uint4*)(bp);     rb.q[1] = *(const uint4*)(bp + 8);
    __syncthreads();
#pragma unroll
    for (int j = 0; j < 8; ++j) {
      *(unsigned int*)(&As[0][lr][lc + 2 * j]) = ra.d[j];
      *(unsigned int*)(&Bs[0][lr][lc + 2 * j]) = rb.d[j];
    }
    __syncthreads();
    const v16bf a0 = load_frag(&As[0][wm][0],      34, lane);
    const v16bf a1 = load_frag(&As[0][wm + 16][0], 34, lane);
#pragma unroll
    for (int ni = 0; ni < 4; ++ni) {
      const v16bf bf = load_frag(&Bs[0][wn + ni * 16][0], 34, lane);
      c[0][ni] = __builtin_amdgcn_wmma_f32_16x16x32_bf16(
          false, a0, false, bf, (short)0, c[0][ni], false, false);
      c[1][ni] = __builtin_amdgcn_wmma_f32_16x16x32_bf16(
          false, a1, false, bf, (short)0, c[1][ni], false, false);
    }
  }
#endif

  const int colh = lane & 15;
  const int rowh = (lane >> 4) * 8;
#pragma unroll
  for (int mi = 0; mi < 2; ++mi)
#pragma unroll
    for (int ni = 0; ni < 4; ++ni) {
      const int col = nbase + wn + ni * 16 + colh;
      const float bv = bias ? bias[col] : 0.f;
#pragma unroll
      for (int r = 0; r < 8; ++r) {
        const int row = mbase + wm + mi * 16 + rowh + r;
        float v = c[mi][ni][r] + bv;
        if (act == 1) v = v > 0.f ? v : 0.01f * v;   // leaky_relu(0.01)
        const size_t off = (size_t)row * N + col;
        if (outF) outF[off] = v;
        if (outB) outB[off] = f2bf(v);
      }
    }
}

// ----------------------------- attention -----------------------------------
// Flash-style softmax(Q K^T / 32) V per (b,h). Q/K/V/O are the contiguous
// raw-reshape buffers: per bh a [512,64] block. One WG per (bh, 128-row qtile).
// Q tile and each K chunk arrive via TDM (pad -> 66-halfword rows); V is
// transposed manually into LDS.
__global__ __launch_bounds__(256) void attn_kernel(
    const unsigned short* __restrict__ Q, const unsigned short* __restrict__ K,
    const unsigned short* __restrict__ V, unsigned short* __restrict__ O) {
  __shared__ unsigned short Qs[128][66];
  __shared__ unsigned short Ks[64][66];
  __shared__ unsigned short Vt[64][66];       // transposed: [dh][key]
  __shared__ unsigned short Ps[8][16][66];    // per-wave P tile staging

  const int t = threadIdx.x;
  const int lane = t & 31;
  const int w = t >> 5;
  const int qt = blockIdx.x & 3;
  const int bh = blockIdx.x >> 2;
  const size_t hb = (size_t)bh * SEQ * DH;

#if HAVE_TDM
  if (w == 0)
    tdm_load_tile(lds_addr_of(&Qs[0][0]), Q + hb + (size_t)qt * 128 * DH,
                  64, 128, DH, 4);
#else
  { // load the 128x64 Q tile
    const int r = t >> 1, ch = (t & 1) * 32;
    union { uint4 q[4]; unsigned int d[16]; } rq;
#pragma unroll
    for (int j = 0; j < 4; ++j)
      rq.q[j] = *(const uint4*)(Q + hb + (size_t)(qt * 128 + r) * DH + ch + 8 * j);
#pragma unroll
    for (int j = 0; j < 16; ++j)
      *(unsigned int*)(&Qs[r][ch + 2 * j]) = rq.d[j];
  }
#endif

  const v8f zero = {0.f, 0.f, 0.f, 0.f, 0.f, 0.f, 0.f, 0.f};
  float rm[8], rl[8];
  v8f oc[4];
#pragma unroll
  for (int r = 0; r < 8; ++r) { rm[r] = -1e30f; rl[r] = 0.f; }
#pragma unroll
  for (int ni = 0; ni < 4; ++ni) oc[ni] = zero;

  for (int kc = 0; kc < SEQ; kc += 64) {
    __syncthreads();                          // prev chunk compute done
#if HAVE_TDM
    if (w == 0)
      tdm_load_tile(lds_addr_of(&Ks[0][0]), K + hb + (size_t)kc * DH,
                    64, 64, DH, 4);
#endif
    { // load V chunk transposed (and K manually on the non-TDM path)
      const int r = t >> 2, ch = (t & 3) * 16;
      union { uint4 q[2]; unsigned int d[8]; unsigned short s[16]; } rv;
      const unsigned short* vp = V + hb + (size_t)(kc + r) * DH + ch;
      rv.q[0] = *(const uint4*)(vp); rv.q[1] = *(const uint4*)(vp + 8);
#if !HAVE_TDM
      union { uint4 q[2]; unsigned int d[8]; } rk;
      const unsigned short* kp = K + hb + (size_t)(kc + r) * DH + ch;
      rk.q[0] = *(const uint4*)(kp); rk.q[1] = *(const uint4*)(kp + 8);
#pragma unroll
      for (int j = 0; j < 8; ++j)
        *(unsigned int*)(&Ks[r][ch + 2 * j]) = rk.d[j];
#endif
#pragma unroll
      for (int j = 0; j < 16; ++j)
        Vt[ch + j][r] = rv.s[j];
    }
#if HAVE_TDM
    if (w == 0) __builtin_amdgcn_s_wait_tensorcnt(0);
#endif
    __syncthreads();

    // scores: 16 q-rows x 64 keys for this wave
    v8f sc[4];
#pragma unroll
    for (int ni = 0; ni < 4; ++ni) sc[ni] = zero;
#pragma unroll
    for (int kb = 0; kb < DH; kb += 32) {
      const v16bf aq = load_frag(&Qs[w * 16][kb], 66, lane);
#pragma unroll
      for (int ni = 0; ni < 4; ++ni) {
        const v16bf bk = load_frag(&Ks[ni * 16][kb], 66, lane);
        sc[ni] = __builtin_amdgcn_wmma_f32_16x16x32_bf16(
            false, aq, false, bk, (short)0, sc[ni], false, false);
      }
    }

    // online softmax (scale = 1/(DH/2) = 1/32); rows split per C-layout:
    // lanes 0-15 hold rows r, lanes 16-31 hold rows r+8 -> 16-lane reductions
#pragma unroll
    for (int r = 0; r < 8; ++r) {
      float mx = -1e30f;
#pragma unroll
      for (int ni = 0; ni < 4; ++ni) {
        sc[ni][r] *= 0.03125f;
        mx = fmaxf(mx, sc[ni][r]);
      }
#pragma unroll
      for (int msk = 8; msk; msk >>= 1) mx = fmaxf(mx, __shfl_xor(mx, msk, 32));
      const float mnew = fmaxf(rm[r], mx);
      const float corr = __expf(rm[r] - mnew);
      float ps = 0.f;
#pragma unroll
      for (int ni = 0; ni < 4; ++ni) {
        const float p = __expf(sc[ni][r] - mnew);
        sc[ni][r] = p;
        ps += p;
      }
#pragma unroll
      for (int msk = 8; msk; msk >>= 1) ps += __shfl_xor(ps, msk, 32);
      rm[r] = mnew;
      rl[r] = rl[r] * corr + ps;
#pragma unroll
      for (int ni = 0; ni < 4; ++ni) oc[ni][r] *= corr;
    }

    // C-layout -> A-layout reshape of P through LDS
#pragma unroll
    for (int ni = 0; ni < 4; ++ni)
#pragma unroll
      for (int r = 0; r < 8; ++r)
        Ps[w][(lane >> 4) * 8 + r][ni * 16 + (lane & 15)] = f2bf(sc[ni][r]);
    __syncthreads();

    // O += P @ V
#pragma unroll
    for (int kb = 0; kb < 64; kb += 32) {
      const v16bf ap = load_frag(&Ps[w][0][kb], 66, lane);
#pragma unroll
      for (int ni = 0; ni < 4; ++ni) {
        const v16bf bv = load_frag(&Vt[ni * 16][kb], 66, lane);
        oc[ni] = __builtin_amdgcn_wmma_f32_16x16x32_bf16(
            false, ap, false, bv, (short)0, oc[ni], false, false);
      }
    }
  }

  // finalize: O /= l, store bf16 ctx (contiguous == raw reshape back)
#pragma unroll
  for (int r = 0; r < 8; ++r) {
    const float inv = 1.0f / rl[r];
    const int row = qt * 128 + w * 16 + (lane >> 4) * 8 + r;
#pragma unroll
    for (int ni = 0; ni < 4; ++ni) {
      const int col = ni * 16 + (lane & 15);
      O[hb + (size_t)row * DH + col] = f2bf(oc[ni][r] * inv);
    }
  }
}

// ------------------------------ VAE head -----------------------------------
__global__ __launch_bounds__(256) void vae_kernel(
    const float* __restrict__ mu, const float* __restrict__ ls,
    const float* __restrict__ eps, float* __restrict__ z, int n) {
  const int i = blockIdx.x * 256 + threadIdx.x;
  if (i < n) z[i] = mu[i] + __expf(ls[i]) * eps[i];
}

// ---------------------------------------------------------------------------
extern "C" void kernel_launch(void* const* d_in, const int* in_sizes, int n_in,
                              void* d_out, int out_size, void* d_ws, size_t ws_size,
                              hipStream_t stream) {
  (void)in_sizes; (void)n_in; (void)out_size; (void)ws_size;
  const int*   x     = (const int*)  d_in[0];
  const float* embed = (const float*)d_in[1];
  const float* Wq  = (const float*)d_in[2];  const float* bq   = (const float*)d_in[3];
  const float* Wk  = (const float*)d_in[4];  const float* bk   = (const float*)d_in[5];
  const float* Wv  = (const float*)d_in[6];  const float* bv   = (const float*)d_in[7];
  const float* Wo  = (const float*)d_in[8];  const float* bo   = (const float*)d_in[9];
  const float* g1  = (const float*)d_in[10]; const float* be1  = (const float*)d_in[11];
  const float* W1  = (const float*)d_in[12]; const float* bf1  = (const float*)d_in[13];
  const float* W2  = (const float*)d_in[14]; const float* bf2  = (const float*)d_in[15];
  const float* g2  = (const float*)d_in[16]; const float* be2  = (const float*)d_in[17];
  const float* g3  = (const float*)d_in[18]; const float* be3  = (const float*)d_in[19];
  const float* Wmu = (const float*)d_in[20]; const float* bmu  = (const float*)d_in[21];
  const float* Wsg = (const float*)d_in[22]; const float* bsg  = (const float*)d_in[23];
  const float* eps = (const float*)d_in[24];
  float* zout = (float*)d_out;

  char* p = (char*)d_ws;
  auto alloc = [&](size_t bytes) -> void* {
    void* r = (void*)p;
    p += (bytes + 255) & ~(size_t)255;
    return r;
  };
  const size_t AB = (size_t)MROWS * DIM;       // activation elements
  unsigned short* hbf   = (unsigned short*)alloc(AB * 2);
  float*          h32   = (float*)         alloc(AB * 4);
  float*          y32   = (float*)         alloc(AB * 4);
  unsigned short* qbf   = (unsigned short*)alloc(AB * 2);
  unsigned short* kbf   = (unsigned short*)alloc(AB * 2);
  unsigned short* vbf   = (unsigned short*)alloc(AB * 2);
  unsigned short* tmpbf = (unsigned short*)alloc(AB * 2);   // ctx, then f1
  unsigned short* wqT  = (unsigned short*)alloc((size_t)DIM * DIM * 2);
  unsigned short* wkT  = (unsigned short*)alloc((size_t)DIM * DIM * 2);
  unsigned short* wvT  = (unsigned short*)alloc((size_t)DIM * DIM * 2);
  unsigned short* woT  = (unsigned short*)alloc((size_t)DIM * DIM * 2);
  unsigned short* w1T  = (unsigned short*)alloc((size_t)DIM * DIM * 2);
  unsigned short* w2T  = (unsigned short*)alloc((size_t)DIM * DIM * 2);
  unsigned short* wmuT = (unsigned short*)alloc((size_t)DIM * LAT * 2);
  unsigned short* wsgT = (unsigned short*)alloc((size_t)DIM * LAT * 2);
  float* mu32 = (float*)alloc((size_t)MROWS * LAT * 4);
  float* ls32 = (float*)alloc((size_t)MROWS * LAT * 4);

  const dim3 blk(256);
  const dim3 gT(DIM / 32, DIM / 32);       // 1024x1024 transposes
  const dim3 gTs(LAT / 32, DIM / 32);      // 1024x128 transposes
  transpose_cvt<<<gT,  blk, 0, stream>>>(Wq,  wqT,  DIM, DIM);
  transpose_cvt<<<gT,  blk, 0, stream>>>(Wk,  wkT,  DIM, DIM);
  transpose_cvt<<<gT,  blk, 0, stream>>>(Wv,  wvT,  DIM, DIM);
  transpose_cvt<<<gT,  blk, 0, stream>>>(Wo,  woT,  DIM, DIM);
  transpose_cvt<<<gT,  blk, 0, stream>>>(W1,  w1T,  DIM, DIM);
  transpose_cvt<<<gT,  blk, 0, stream>>>(W2,  w2T,  DIM, DIM);
  transpose_cvt<<<gTs, blk, 0, stream>>>(Wmu, wmuT, DIM, LAT);
  transpose_cvt<<<gTs, blk, 0, stream>>>(Wsg, wsgT, DIM, LAT);

  embed_ln_kernel<<<MROWS, blk, 0, stream>>>(x, embed, g1, be1, h32, hbf);

  const dim3 gG(DIM / 128, MROWS / 128);   // (8, 64)
  const dim3 gL(LAT / 128, MROWS / 128);   // (1, 64)
  gemm_bf16<<<gG, blk, 0, stream>>>(hbf, wqT, bq, nullptr, qbf, MROWS, DIM, DIM, 0);
  gemm_bf16<<<gG, blk, 0, stream>>>(hbf, wkT, bk, nullptr, kbf, MROWS, DIM, DIM, 0);
  gemm_bf16<<<gG, blk, 0, stream>>>(hbf, wvT, bv, nullptr, vbf, MROWS, DIM, DIM, 0);

  attn_kernel<<<16 * 16 * (SEQ / 128), blk, 0, stream>>>(qbf, kbf, vbf, tmpbf);

  gemm_bf16<<<gG, blk, 0, stream>>>(tmpbf, woT, bo, y32, nullptr, MROWS, DIM, DIM, 0);
  resid_ln_kernel<<<MROWS, blk, 0, stream>>>(y32, h32, g2, be2, h32, hbf);

  gemm_bf16<<<gG, blk, 0, stream>>>(hbf, w1T, bf1, nullptr, tmpbf, MROWS, DIM, DIM, 1);
  gemm_bf16<<<gG, blk, 0, stream>>>(tmpbf, w2T, bf2, y32, nullptr, MROWS, DIM, DIM, 0);
  resid_ln_kernel<<<MROWS, blk, 0, stream>>>(y32, y32, g3, be3, h32, hbf);  // f + f

  gemm_bf16<<<gL, blk, 0, stream>>>(hbf, wmuT, bmu, mu32, nullptr, MROWS, LAT, DIM, 0);
  gemm_bf16<<<gL, blk, 0, stream>>>(hbf, wsgT, bsg, ls32, nullptr, MROWS, LAT, DIM, 0);

  const int n = MROWS * LAT;
  vae_kernel<<<(n + 255) / 256, blk, 0, stream>>>(mu32, ls32, eps, zout, n);
}
